// ContextRCNN_50800873177169
// MI455X (gfx1250) — matrix operand order repaired and checked by reading, not compiled, not run on hardware
//
#include <hip/hip_runtime.h>
#include <hip/hip_bf16.h>

typedef _Float16 v16h __attribute__((ext_vector_type(16)));
typedef _Float16 h8   __attribute__((ext_vector_type(8)));
typedef float    v8f  __attribute__((ext_vector_type(8)));

#define SOFTMAX_SCALE 6.25f   // 1/(0.01*sqrt(256))

// ---------- helpers ----------

__device__ __forceinline__ v16h cat16(h8 a, h8 b) {
  return __builtin_shufflevector(a, b, 0, 1, 2, 3, 4, 5, 6, 7,
                                 8, 9, 10, 11, 12, 13, 14, 15);
}

// 16 consecutive halves at p (two 16B loads) -> B operand
__device__ __forceinline__ v16h load16(const _Float16* p) {
  return cat16(*(const h8*)p, *(const h8*)(p + 8));
}

// 8 halves at p and 8 halves at p+8 -> A operand (K = base+0..15)
__device__ __forceinline__ v16h load8x2(const _Float16* p) {
  return cat16(*(const h8*)p, *(const h8*)(p + 8));
}

__device__ __forceinline__ v8f wmma_f16(v16h a, v16h b, v8f c) {
  return __builtin_amdgcn_wmma_f32_16x16x32_f16(false, a, false, b,
                                                (short)0, c, false, false);
}

// ---------- 1) spatial mean pool [rows, 7*7] f32 -> [rows] f16 ----------
// one wave per (n,c) row; 49 contiguous floats; non-temporal (read-once stream).
__global__ void pool_kernel(const float* __restrict__ x, _Float16* __restrict__ out,
                            int nrows) {
  int wid  = blockIdx.x * (blockDim.x >> 5) + (threadIdx.x >> 5);
  int lane = threadIdx.x & 31;
  if (wid >= nrows) return;
  const float* p = x + (size_t)wid * 49;
  float v = __builtin_nontemporal_load(p + lane);
  if (lane + 32 < 49) v += __builtin_nontemporal_load(p + lane + 32);
#pragma unroll
  for (int m = 16; m >= 1; m >>= 1) v += __shfl_xor(v, m, 32);
  if (lane == 0) out[wid] = (_Float16)(v * (1.0f / 49.0f));
}

// ---------- 2) weight transpose+convert:  W[K,N] f32 -> Wt[N,K] f16 ----------
__global__ void wtrans_kernel(const float* __restrict__ W, _Float16* __restrict__ out,
                              int K, int N) {
  int idx = blockIdx.x * blockDim.x + threadIdx.x;
  if (idx >= K * N) return;
  int k = idx / N, n = idx - k * N;
  out[(size_t)n * K + k] = (_Float16)W[idx];
}

// ---------- 3) WMMA GEMM:  out = act(A[M,K](f16) @ Bt[N,K](f16, pre-transposed)) ----------
// 4 waves/block, each wave one 16x16 tile; all shapes compile-time.
// MODE: 0 = f32 row-major, 1 = f16 row-major, 2 = f16 transposed [N,M]
template <int M, int K, int N, int MODE, int RELU>
__global__ __launch_bounds__(128)
void wmma_gemm_kernel(const _Float16* __restrict__ A, const _Float16* __restrict__ Bt,
                      void* __restrict__ out) {
  const int lane    = threadIdx.x & 31;
  const int wv      = threadIdx.x >> 5;
  const int m0      = blockIdx.y << 4;
  const int n0      = (blockIdx.x * 4 + wv) << 4;
  const int half_hi = lane >> 4;
  const int abase   = half_hi ? 16 : 0;
  const int kb2     = half_hi ? 16 : 0;
  const int ncol    = n0 + (lane & 15);

  const _Float16* arow = A + (m0 + (lane & 15)) * K + abase;
  const _Float16* brow = Bt + ncol * K + kb2;
  v8f acc = {};
#pragma unroll 4
  for (int kc = 0; kc < K; kc += 32) {
    v16h a = load8x2(arow + kc);
    v16h b = load16(brow + kc);
    acc = wmma_f16(a, b, acc);
  }
  const int rbase = half_hi ? 8 : 0;
#pragma unroll
  for (int r = 0; r < 8; ++r) {
    int row = m0 + rbase + r;
    float v = acc[r];
    if (RELU) v = fmaxf(v, 0.0f);
    if constexpr (MODE == 0)      ((float*)out)[row * N + ncol] = v;
    else if constexpr (MODE == 1) ((_Float16*)out)[row * N + ncol] = (_Float16)v;
    else                          ((_Float16*)out)[ncol * M + row] = (_Float16)v;
  }
}

// ---------- 4) row-wise L2 normalize [rows,256] f32 -> f16 ----------
__global__ void l2norm_kernel(const float* __restrict__ x, _Float16* __restrict__ out,
                              int nrows) {
  int wid  = blockIdx.x * (blockDim.x >> 5) + (threadIdx.x >> 5);
  int lane = threadIdx.x & 31;
  if (wid >= nrows) return;
  const float* p = x + (size_t)wid * 256;
  float ss = 0.0f;
#pragma unroll
  for (int i = 0; i < 8; ++i) { float t = p[lane + 32 * i]; ss += t * t; }
#pragma unroll
  for (int m = 16; m >= 1; m >>= 1) ss += __shfl_xor(ss, m, 32);
  float inv = 1.0f / fmaxf(sqrtf(ss), 1e-12f);
  _Float16* o = out + (size_t)wid * 256;
#pragma unroll
  for (int i = 0; i < 8; ++i) o[lane + 32 * i] = (_Float16)(p[lane + 32 * i] * inv);
}

// ---------- 5) fused flash attention ----------
// Q[4096,256] f16, Kf[T,256] f16, Vt[256,T] f16 (transposed) -> Out[4096,256] f16
// block = 2 waves; Q tile staged once in LDS (shared), each wave owns 128 output feats.
__global__ __launch_bounds__(64)
void attention_kernel(const _Float16* __restrict__ Q, const _Float16* __restrict__ Kf,
                      const _Float16* __restrict__ Vt, _Float16* __restrict__ Out,
                      int T) {
  __shared__ __align__(16) _Float16 Qlds[16][256];    // 8 KB, shared by both waves
  __shared__ __align__(16) _Float16 P_lds[2][16][32]; // per-wave P tiles
  const int tid     = threadIdx.x;
  const int wv      = tid >> 5;
  const int lane    = tid & 31;
  const int m0      = blockIdx.x << 4;
  const int half_hi = lane >> 4;
  const int nn      = lane & 15;
  const int abase   = half_hi ? 16 : 0;
  const int kb2     = half_hi ? 16 : 0;
  const int f0      = wv * 128;

  // cooperative Q tile load: 16x256 f16 = 512 x 16B, coalesced
  {
    const uint4* g = (const uint4*)(Q + (size_t)m0 * 256);
    uint4* s = (uint4*)&Qlds[0][0];
#pragma unroll
    for (int i = 0; i < 8; ++i) s[i * 64 + tid] = g[i * 64 + tid];
  }
  __syncthreads();

  v8f o[8];
#pragma unroll
  for (int t = 0; t < 8; ++t) o[t] = (v8f){};
  float mrow[8], lrow[8];
#pragma unroll
  for (int r = 0; r < 8; ++r) { mrow[r] = -1e30f; lrow[r] = 0.0f; }

  const _Float16* vbase0 = Vt + (size_t)(f0 + nn) * T + kb2;

  for (int t0 = 0; t0 < T; t0 += 32) {
    // scores S[16 x 32] = Q @ K^T  (two 16x16 C/D tiles), double-buffered B loads
    const _Float16* kbp0 = Kf + (size_t)(t0 + nn)      * 256 + kb2;
    const _Float16* kbp1 = Kf + (size_t)(t0 + 16 + nn) * 256 + kb2;
    v8f s0 = (v8f){}, s1 = (v8f){};
    v16h b0 = load16(kbp0), b1 = load16(kbp1);
#pragma unroll
    for (int c = 0; c < 8; ++c) {
      v16h nb0, nb1;
      if (c < 7) { nb0 = load16(kbp0 + (c + 1) * 32); nb1 = load16(kbp1 + (c + 1) * 32); }
      v16h qa = load8x2(&Qlds[nn][c * 32 + abase]);   // A operand from LDS
      s0 = wmma_f16(qa, b0, s0);
      s1 = wmma_f16(qa, b1, s1);
      b0 = nb0; b1 = nb1;
    }
    // online softmax per row (row of 16 lives in one VGPR across a 16-lane half)
#pragma unroll
    for (int r = 0; r < 8; ++r) {
      float a0 = s0[r] * SOFTMAX_SCALE, a1 = s1[r] * SOFTMAX_SCALE;
      float mx = fmaxf(a0, a1);
      mx = fmaxf(mx, __shfl_xor(mx, 1, 32));
      mx = fmaxf(mx, __shfl_xor(mx, 2, 32));
      mx = fmaxf(mx, __shfl_xor(mx, 4, 32));
      mx = fmaxf(mx, __shfl_xor(mx, 8, 32));
      float mnew  = fmaxf(mrow[r], mx);
      float alpha = __expf(mrow[r] - mnew);
      float p0 = __expf(a0 - mnew);
      float p1 = __expf(a1 - mnew);
      float rs = p0 + p1;
      rs += __shfl_xor(rs, 1, 32);
      rs += __shfl_xor(rs, 2, 32);
      rs += __shfl_xor(rs, 4, 32);
      rs += __shfl_xor(rs, 8, 32);
      lrow[r] = lrow[r] * alpha + rs;
      mrow[r] = mnew;
#pragma unroll
      for (int t = 0; t < 8; ++t) o[t][r] *= alpha;
      int m = r + (half_hi ? 8 : 0);
      P_lds[wv][m][nn]      = (_Float16)p0;
      P_lds[wv][m][16 + nn] = (_Float16)p1;
    }
    __syncthreads();
    // re-read P in A layout and accumulate O += P @ V, double-buffered V loads
    v16h pa = load8x2(&P_lds[wv][nn][abase]);
    const _Float16* vb0 = vbase0 + t0;
    v16h vb = load16(vb0);
#pragma unroll
    for (int t = 0; t < 8; ++t) {
      v16h nvb;
      if (t < 7) nvb = load16(vb0 + (size_t)(t + 1) * 16 * T);
      o[t] = wmma_f16(pa, vb, o[t]);
      vb = nvb;
    }
    __syncthreads();
  }
  // normalize and store f16 row-major [4096,256]
#pragma unroll
  for (int r = 0; r < 8; ++r) {
    float inv = 1.0f / lrow[r];
    int row = m0 + r + (half_hi ? 8 : 0);
#pragma unroll
    for (int t = 0; t < 8; ++t)
      Out[(size_t)row * 256 + f0 + t * 16 + nn] = (_Float16)(o[t][r] * inv);
  }
}

// ---------------------------- host launcher ----------------------------
extern "C" void kernel_launch(void* const* d_in, const int* in_sizes, int n_in,
                              void* d_out, int out_size, void* d_ws, size_t ws_size,
                              hipStream_t stream) {
  (void)in_sizes; (void)n_in; (void)out_size; (void)ws_size;
  const float* central = (const float*)d_in[0];
  const float* context = (const float*)d_in[1];
  const float* qw1 = (const float*)d_in[2];
  const float* qw2 = (const float*)d_in[3];
  const float* kw1 = (const float*)d_in[4];
  const float* kw2 = (const float*)d_in[5];
  const float* vw1 = (const float*)d_in[6];
  const float* vw2 = (const float*)d_in[7];
  const float* fw1 = (const float*)d_in[8];
  const float* fw2 = (const float*)d_in[9];

  const int N = 4096, T = 12288, C = 256;

  // workspace layout (~50 MB)
  char* ws = (char*)d_ws;
  _Float16* poolC  = (_Float16*)(ws);                       // [4096,256]  2 MB
  _Float16* poolX  = (_Float16*)(ws + (2u   << 20));        // [12288,256] 6 MB
  _Float16* hbuf   = (_Float16*)(ws + (8u   << 20));        // [<=12288,512] 12 MB
  float*    rawbuf = (float*)   (ws + (20u  << 20));        // [<=12288,256] 12 MB
  _Float16* q_h    = (_Float16*)(ws + (32u  << 20));        // [4096,256]  2 MB
  _Float16* k_h    = (_Float16*)(ws + (34u  << 20));        // [12288,256] 6 MB
  _Float16* vT_h   = (_Float16*)(ws + (40u  << 20));        // [256,12288] 6 MB
  _Float16* attn_h = (_Float16*)(ws + (46u  << 20));        // [4096,256]  2 MB
  // transposed f16 weights, 256 KB each (131072 elements)
  _Float16* qw1t = (_Float16*)(ws + (48u << 20));
  _Float16* qw2t = qw1t + 131072;
  _Float16* kw1t = qw2t + 131072;
  _Float16* kw2t = kw1t + 131072;
  _Float16* vw1t = kw2t + 131072;
  _Float16* vw2t = vw1t + 131072;
  _Float16* fw1t = vw2t + 131072;
  _Float16* fw2t = fw1t + 131072;

  // one-time weight transpose+convert to f16 [N,K] (6 MB total traffic)
  wtrans_kernel<<<512, 256, 0, stream>>>(qw1, qw1t, 256, 512);
  wtrans_kernel<<<512, 256, 0, stream>>>(qw2, qw2t, 512, 256);
  wtrans_kernel<<<512, 256, 0, stream>>>(kw1, kw1t, 256, 512);
  wtrans_kernel<<<512, 256, 0, stream>>>(kw2, kw2t, 512, 256);
  wtrans_kernel<<<512, 256, 0, stream>>>(vw1, vw1t, 256, 512);
  wtrans_kernel<<<512, 256, 0, stream>>>(vw2, vw2t, 512, 256);
  wtrans_kernel<<<512, 256, 0, stream>>>(fw1, fw1t, 256, 512);
  wtrans_kernel<<<512, 256, 0, stream>>>(fw2, fw2t, 512, 256);

  // mean pools (the only pass over the 822 MB of input)
  pool_kernel<<<(N * C + 7) / 8, 256, 0, stream>>>(central, poolC, N * C);
  pool_kernel<<<(T * C + 7) / 8, 256, 0, stream>>>(context, poolX, T * C);

  // queries = l2norm(relu(poolC@qw1)@qw2)
  wmma_gemm_kernel<4096, 256, 512, 1, 1>
      <<<dim3(512 / 64, 4096 / 16), 128, 0, stream>>>(poolC, qw1t, hbuf);
  wmma_gemm_kernel<4096, 512, 256, 0, 0>
      <<<dim3(256 / 64, 4096 / 16), 128, 0, stream>>>(hbuf, qw2t, rawbuf);
  l2norm_kernel<<<N / 8, 256, 0, stream>>>(rawbuf, q_h, N);

  // keys = l2norm(relu(poolX@kw1)@kw2)
  wmma_gemm_kernel<12288, 256, 512, 1, 1>
      <<<dim3(512 / 64, 12288 / 16), 128, 0, stream>>>(poolX, kw1t, hbuf);
  wmma_gemm_kernel<12288, 512, 256, 0, 0>
      <<<dim3(256 / 64, 12288 / 16), 128, 0, stream>>>(hbuf, kw2t, rawbuf);
  l2norm_kernel<<<T / 8, 256, 0, stream>>>(rawbuf, k_h, T);

  // values (stored transposed [256,T] so attention B-loads are contiguous)
  wmma_gemm_kernel<12288, 256, 512, 1, 1>
      <<<dim3(512 / 64, 12288 / 16), 128, 0, stream>>>(poolX, vw1t, hbuf);
  wmma_gemm_kernel<12288, 512, 256, 2, 0>
      <<<dim3(256 / 64, 12288 / 16), 128, 0, stream>>>(hbuf, vw2t, vT_h);

  // fused softmax attention: attn = softmax(q k^T * 6.25) @ v
  attention_kernel<<<N / 16, 64, 0, stream>>>(q_h, k_h, vT_h, attn_h, T);

  // final mapper -> f32 output
  wmma_gemm_kernel<4096, 256, 512, 1, 1>
      <<<dim3(512 / 64, 4096 / 16), 128, 0, stream>>>(attn_h, fw1t, hbuf);
  wmma_gemm_kernel<4096, 512, 256, 0, 0>
      <<<dim3(256 / 64, 4096 / 16), 128, 0, stream>>>(hbuf, fw2t, (float*)d_out);
}
